// GRUModel_21715354648612
// MI455X (gfx1250) — matrix operand order repaired and checked by reading, not compiled, not run on hardware
//
#include <hip/hip_runtime.h>
#include <hip/hip_bf16.h>
#include <hip/hip_fp16.h>

typedef __attribute__((ext_vector_type(16))) _Float16 v16h;
typedef __attribute__((ext_vector_type(8)))  _Float16 v8h;
typedef __attribute__((ext_vector_type(8)))  float    v8f;
typedef __attribute__((ext_vector_type(4)))  float    v4f;

#define B_TOT 4096
#define T_LEN 512
#define IN_D  64
#define H_D   128
#define G3    384      // 3*H

#define SWX 72         // padded halfs per Wx row (64 + 8)  -> bank rotation
#define SWH 136        // padded halfs per Wh row (128 + 8) -> bank rotation
#define SHS 136        // h staging row stride in halfs

#define WAVES 8        // 4 wave-pairs
#define BLK   (WAVES * 32)
#define TILES 4        // batch tiles (16 rows each) per block
#define SBUF  (TILES * 16 * SHS)   // one h staging buffer (halfs)

union V16U { v16h v; v8h h[2]; };

// A-fragment (16x32 f16): lanep encodes row*stride + (lane>=16 ? 8 : 0).
__device__ inline v16h fragA(const _Float16* lanep, int kb) {
  V16U u;
  u.h[0] = *(const v8h*)(lanep + kb);
  u.h[1] = *(const v8h*)(lanep + kb + 16);
  return u.v;
}

// B-fragment (32x16 f16): lanep encodes col*stride + (lane>=16 ? 8 : 0) (+ q*64 rows);
// off is a compile-time constant after unroll.
__device__ inline v16h fragB(const _Float16* lanep, int off) {
  V16U u;
  u.h[0] = *(const v8h*)(lanep + off);
  u.h[1] = *(const v8h*)(lanep + off + 16);
  return u.v;
}

__device__ inline v8f wmma32(v16h a, v16h b, v8f c) {
  return __builtin_amdgcn_wmma_f32_16x16x32_f16(false, a, false, b, (short)0, c, false, false);
}

extern "C" __global__ void __launch_bounds__(BLK, 1)
gru_wmma_kernel(const float* __restrict__ seq, const float* __restrict__ Wx,
                const float* __restrict__ Wh, const float* __restrict__ bias,
                const float* __restrict__ fcw, const float* __restrict__ fcb,
                float* __restrict__ out)
{
  extern __shared__ _Float16 smem[];
  _Float16* wx = smem;                          // 384*72 halfs
  _Float16* wh = wx + G3 * SWX;                 // 384*136 halfs
  _Float16* hs = wh + G3 * SWH;                 // 2 * SBUF halfs (ping-pong)
  float*    red = (float*)(hs + 2 * SBUF);      // WAVES * 32*8 floats

  const int tid  = threadIdx.x;
  const int wave = tid >> 5;
  const int lane = tid & 31;
  const int col  = lane & 15;                   // N coordinate in fragment layouts
  const int hi   = lane >> 4;                   // upper half-wave flag
  const int pair = wave >> 1;                   // batch-tile index within block
  const int q    = wave & 1;                    // gate-column half (j = q*4 .. q*4+3)

  // ---- stage weights f32 -> f16 into padded LDS; zero both h buffers ----
  for (int i = tid; i < G3 * IN_D; i += BLK) {
    int r = i >> 6, c = i & 63;
    wx[r * SWX + c] = (_Float16)Wx[i];
  }
  for (int i = tid; i < G3 * H_D; i += BLK) {
    int r = i >> 7, c = i & 127;
    wh[r * SWH + c] = (_Float16)Wh[i];
  }
  for (int i = tid; i < 2 * SBUF; i += BLK) hs[i] = (_Float16)0.f;
  __syncthreads();

  // ---- per-lane gate biases for this wave's 4 column tiles ----
  const int jbase = q * 4;
  float br[4], bz[4], bh[4];
  #pragma unroll
  for (int jj = 0; jj < 4; ++jj) {
    int idx = (jbase + jj) * 16 + col;
    br[jj] = bias[        idx];
    bz[jj] = bias[H_D   + idx];
    bh[jj] = bias[2*H_D + idx];
  }

  // this wave's 4 h column-tiles, resident in fp32 C-fragments
  v8f hacc[4];
  #pragma unroll
  for (int jj = 0; jj < 4; ++jj) { v8f z = {}; hacc[jj] = z; }

  const int b0   = blockIdx.x * (TILES * 16) + pair * 16;
  const int arow = col;                         // A-layout M for this lane
  const float* xrow = seq + (size_t)(b0 + arow) * T_LEN * IN_D;
  const int xo = hi ? 8 : 0;

  // per-lane LDS base pointers; q*64 gate-row offset folded in -> const offsets
  const _Float16* wxq = wx + (q * 64) * SWX + col * SWX + (hi << 3);
  const _Float16* whq = wh + (q * 64) * SWH + col * SWH + (hi << 3);
  const int pairBase = pair * 16 * SHS;

  #pragma unroll 1
  for (int t = 0; t < T_LEN; ++t) {
    const int rb = t & 1, wb = rb ^ 1;
    const _Float16* ahp = hs + rb * SBUF + pairBase + arow * SHS + (hi << 3);
    _Float16*       stp = hs + wb * SBUF + pairBase + (hi ? 8 * SHS : 0) + q * 64 + col;

    const float* xp = xrow + (size_t)t * IN_D;
    __builtin_prefetch(xp + IN_D, 0, 1);        // next step's x line

    // ---- x A-fragments: 2 slabs of K=32 from HBM, cvt f32->f16 ----
    v16h ax[2];
    #pragma unroll
    for (int s = 0; s < 2; ++s) {
      v4f f0 = *(const v4f*)(xp + 32 * s + xo);
      v4f f1 = *(const v4f*)(xp + 32 * s + xo + 4);
      v4f g0 = *(const v4f*)(xp + 32 * s + xo + 16);
      v4f g1 = *(const v4f*)(xp + 32 * s + xo + 20);
      v16h a;
      #pragma unroll
      for (int e = 0; e < 4; ++e) {
        a[e]      = (_Float16)f0[e];
        a[4 + e]  = (_Float16)f1[e];
        a[8 + e]  = (_Float16)g0[e];
        a[12 + e] = (_Float16)g1[e];
      }
      ax[s] = a;
    }

    // ---- h A-fragments: 4 slabs of K=32 from ping buffer ----
    v16h ah[4];
    #pragma unroll
    for (int s = 0; s < 4; ++s) ah[s] = fragA(ahp, 32 * s);

    // ---- this wave's 4 gate N-tiles: 72 WMMAs / step ----
    #pragma unroll
    for (int jj = 0; jj < 4; ++jj) {
      v8f accr = {}, accz = {}, acci = {}, acchh = {};
      #pragma unroll
      for (int s = 0; s < 2; ++s) {
        accr = wmma32(ax[s], fragB(wxq, (          jj * 16) * SWX + 32 * s), accr);
        accz = wmma32(ax[s], fragB(wxq, (H_D     + jj * 16) * SWX + 32 * s), accz);
        acci = wmma32(ax[s], fragB(wxq, (2*H_D   + jj * 16) * SWX + 32 * s), acci);
      }
      #pragma unroll
      for (int s = 0; s < 4; ++s) {
        accr  = wmma32(ah[s], fragB(whq, (        jj * 16) * SWH + 32 * s), accr);
        accz  = wmma32(ah[s], fragB(whq, (H_D   + jj * 16) * SWH + 32 * s), accz);
        acchh = wmma32(ah[s], fragB(whq, (2*H_D + jj * 16) * SWH + 32 * s), acchh);
      }

      // gate math in fp32 on C-fragment layout; update h; stage f16 into pong
      v8f h = hacc[jj];
      #pragma unroll
      for (int e = 0; e < 8; ++e) {
        float r   = 1.f / (1.f + __expf(-(accr[e] + br[jj])));
        float z   = 1.f / (1.f + __expf(-(accz[e] + bz[jj])));
        float pre = acci[e] + r * acchh[e] + bh[jj];
        float u   = __expf(-2.f * pre);
        float n   = (1.f - u) / (1.f + u);        // tanh(pre)
        float hv  = n + z * (h[e] - n);
        h[e] = hv;
        stp[e * SHS + jj * 16] = (_Float16)hv;    // row (hi?8:0)+e, col q*64+jj*16+col
      }
      hacc[jj] = h;
    }

    __syncthreads();   // publish pong buffer for step t+1 (pair exchange)
  }

  // ---- final FC (H -> 1): per-lane partial dot + cross-wave reduce ----
  float fw[4];
  #pragma unroll
  for (int jj = 0; jj < 4; ++jj) fw[jj] = fcw[(jbase + jj) * 16 + col];

  float part[8];
  #pragma unroll
  for (int e = 0; e < 8; ++e) part[e] = 0.f;
  #pragma unroll
  for (int jj = 0; jj < 4; ++jj)
    #pragma unroll
    for (int e = 0; e < 8; ++e) part[e] += hacc[jj][e] * fw[jj];

  float* myred = red + wave * 32 * 8;
  #pragma unroll
  for (int e = 0; e < 8; ++e) myred[lane * 8 + e] = part[e];
  __syncthreads();

  if (q == 0 && lane < 16) {
    const float* r0 = red + (2 * pair) * 256;     // this pair's two wave regions
    const float* r1 = r0 + 256;
    float s = 0.f;
    if (lane < 8) {
      for (int L = 0;  L < 16; ++L) s += r0[L * 8 + lane] + r1[L * 8 + lane];
    } else {
      for (int L = 16; L < 32; ++L) s += r0[L * 8 + (lane - 8)] + r1[L * 8 + (lane - 8)];
    }
    out[b0 + lane] = s + fcb[0];
  }
}

extern "C" void kernel_launch(void* const* d_in, const int* in_sizes, int n_in,
                              void* d_out, int out_size, void* d_ws, size_t ws_size,
                              hipStream_t stream) {
  (void)in_sizes; (void)n_in; (void)out_size; (void)d_ws; (void)ws_size;
  const float* seq = (const float*)d_in[0];
  const float* Wx  = (const float*)d_in[1];
  const float* Wh  = (const float*)d_in[2];
  const float* b   = (const float*)d_in[3];
  const float* fcw = (const float*)d_in[4];
  const float* fcb = (const float*)d_in[5];
  float* out = (float*)d_out;

  size_t smem = (size_t)(G3 * SWX + G3 * SWH + 2 * SBUF) * 2   // f16 halfs
              + (size_t)WAVES * 32 * 8 * sizeof(float);        // reduce scratch
  dim3 grid(B_TOT / (TILES * 16));   // 64 blocks, 8 waves, 64 batch rows/block
  gru_wmma_kernel<<<grid, BLK, smem, stream>>>(seq, Wx, Wh, b, fcw, fcb, out);
}
